// VectorQuantizer_30365418782927
// MI455X (gfx1250) — matrix operand order repaired and checked by reading, not compiled, not run on hardware
//
#include <hip/hip_runtime.h>
#include <hip/hip_bf16.h>

typedef __attribute__((ext_vector_type(16))) __bf16 v16bf;
typedef __attribute__((ext_vector_type(8)))  float  v8f;

#define NUM_EMB  512
#define EMB_DIM  64
#define N_ROWS   (64 * 4096)

// ---------------------------------------------------------------------------
// Prep: split codebook into bf16 hi/lo parts, compute ||e_k||^2 (exact fp32)
// ---------------------------------------------------------------------------
__global__ __launch_bounds__(512) void vq_prep(const float* __restrict__ emb,
                                               __bf16* __restrict__ ehi,
                                               __bf16* __restrict__ elo,
                                               float* __restrict__ enorm) {
  int k = threadIdx.x;
  if (k >= NUM_EMB) return;
  float acc = 0.f;
  for (int d = 0; d < EMB_DIM; ++d) {
    float e = emb[k * EMB_DIM + d];
    __bf16 h = (__bf16)e;
    ehi[k * EMB_DIM + d] = h;
    elo[k * EMB_DIM + d] = (__bf16)(e - (float)h);
    acc = fmaf(e, e, acc);
  }
  enorm[k] = acc;
}

// ---------------------------------------------------------------------------
// Main: nearest-code search via bf16-split WMMA, gather, loss partials.
// One wave per 16-row tile; 8 waves per block -> 128 rows per block.
// Loss identity: sum_row (q-x)^2 = ||x_row||^2 + best_score_row, so x is
// read exactly once from HBM.
// ---------------------------------------------------------------------------
__global__ __launch_bounds__(256) void vq_main(const float* __restrict__ x,
                                               const float* __restrict__ emb,
                                               const __bf16* __restrict__ ehi_g,
                                               const __bf16* __restrict__ elo_g,
                                               const float* __restrict__ enorm,
                                               float* __restrict__ out_idx,
                                               float* __restrict__ out_q,
                                               float* __restrict__ partials) {
  __shared__ __bf16 sEhi[NUM_EMB * EMB_DIM];  // 64 KB, codebook hi part

  const int tid = threadIdx.x;

  // Stage codebook-hi into LDS (4096 x 16B, coalesced)
  {
    const uint4* src = (const uint4*)ehi_g;
    uint4* dst = (uint4*)sEhi;
#pragma unroll
    for (int i = 0; i < (NUM_EMB * EMB_DIM * 2 / 16) / 256; ++i)
      dst[tid + i * 256] = src[tid + i * 256];
  }
  __syncthreads();

  const int wave = tid >> 5;
  const int lane = tid & 31;
  const int M = lane & 15;       // row within tile (A) / column (B,C,D)
  const int hi16 = lane >> 4;    // which 16-lane half
  const long row0 = ((long)blockIdx.x * 8 + wave) * 16;

  // ---- Build A operands (16x32 bf16, contiguous K order k = 16*hi16 + s), hi + lo split ----
  // Also accumulate this lane's partial ||x_row||^2 in exact fp32.
  v16bf Ahi[2], Alo[2];
  float sq = 0.f;
  const float* xrow = x + (row0 + M) * EMB_DIM;
#pragma unroll
  for (int h = 0; h < 2; ++h) {            // two K-halves of D=64
    const int base0 = 32 * h + 16 * hi16;  // VGPR0-3 chunk: K = base0..base0+7
    const int base1 = base0 + 8;           // VGPR4-7 chunk: K = base0+8..base0+15
#pragma unroll
    for (int half = 0; half < 2; ++half) {
      const float* p = xrow + (half ? base1 : base0);
#pragma unroll
      for (int j = 0; j < 8; ++j) {
        float f = p[j];
        __bf16 hv = (__bf16)f;
        Ahi[h][half * 8 + j] = hv;
        Alo[h][half * 8 + j] = (__bf16)(f - (float)hv);
        sq = fmaf(f, f, sq);
      }
    }
  }

  float best[8];
  int bidx[8];
#pragma unroll
  for (int r = 0; r < 8; ++r) { best[r] = 3.4e38f; bidx[r] = 0; }

  // ---- Sweep 32 code tiles of 16 codes each (2 independent WMMA chains) ----
  const int kb = hi16 * 16;  // B layout: lane half selects K range
#pragma unroll 2
  for (int n = 0; n < NUM_EMB / 16; ++n) {
    const int code = n * 16 + M;
    const v16bf Bhi0 = *(const v16bf*)&sEhi[code * EMB_DIM + kb];
    const v16bf Bhi1 = *(const v16bf*)&sEhi[code * EMB_DIM + 32 + kb];
    const v16bf Blo0 = *(const v16bf*)(elo_g + code * EMB_DIM + kb);
    const v16bf Blo1 = *(const v16bf*)(elo_g + code * EMB_DIM + 32 + kb);

    v8f acc0 = {}, acc1 = {};
    acc0 = __builtin_amdgcn_wmma_f32_16x16x32_bf16(false, Ahi[0], false, Bhi0, (short)0, acc0, false, false);
    acc1 = __builtin_amdgcn_wmma_f32_16x16x32_bf16(false, Ahi[1], false, Bhi1, (short)0, acc1, false, false);
    acc0 = __builtin_amdgcn_wmma_f32_16x16x32_bf16(false, Alo[0], false, Bhi0, (short)0, acc0, false, false);
    acc1 = __builtin_amdgcn_wmma_f32_16x16x32_bf16(false, Alo[1], false, Bhi1, (short)0, acc1, false, false);
    acc0 = __builtin_amdgcn_wmma_f32_16x16x32_bf16(false, Ahi[0], false, Blo0, (short)0, acc0, false, false);
    acc1 = __builtin_amdgcn_wmma_f32_16x16x32_bf16(false, Ahi[1], false, Blo1, (short)0, acc1, false, false);

    const float en = enorm[code];
#pragma unroll
    for (int r = 0; r < 8; ++r) {
      float s = fmaf(-2.f, acc0[r] + acc1[r], en);  // ||e||^2 - 2 x.e
      if (s < best[r]) { best[r] = s; bidx[r] = code; }
    }
  }

  // ---- Cross-lane argmin within each 16-lane half (C/D layout) ----
#pragma unroll
  for (int off = 1; off <= 8; off <<= 1) {
#pragma unroll
    for (int r = 0; r < 8; ++r) {
      float ov = __shfl_xor(best[r], off, 32);
      int oi = __shfl_xor(bidx[r], off, 32);
      if (ov < best[r] || (ov == best[r] && oi < bidx[r])) {
        best[r] = ov;
        bidx[r] = oi;
      }
    }
  }
  // lanes 0-15 hold min for rows M=r; lanes 16-31 for rows M=r+8

  // ---- Gather codebook rows (L2-hot), write quantized + indices ----
#pragma unroll
  for (int rr = 0; rr < 16; ++rr) {
    int idx = __shfl(bidx[rr & 7], (rr >> 3) << 4, 32);
    long row = row0 + rr;
    float2 q = ((const float2*)(emb + (long)idx * EMB_DIM))[lane];
    ((float2*)(out_q + row * EMB_DIM))[lane] = q;
    if (lane == 0) out_idx[row] = (float)idx;
  }

  // ---- Wave loss: sum_rows(||x||^2 + best).  best duplicated across each
  //      16-lane half, so only lanes 0 and 16 contribute their sums. ----
  float bsum = 0.f;
#pragma unroll
  for (int r = 0; r < 8; ++r) bsum += best[r];
  float contrib = sq + ((lane & 15) == 0 ? bsum : 0.f);
#pragma unroll
  for (int off = 16; off >= 1; off >>= 1) contrib += __shfl_xor(contrib, off, 32);

  __syncthreads();                       // everyone is done reading sEhi
  float* sL = (float*)sEhi;              // reuse LDS for block loss reduce
  if (lane == 0) sL[wave] = contrib;
  __syncthreads();
  if (tid == 0) {
    float t = 0.f;
#pragma unroll
    for (int w = 0; w < 8; ++w) t += sL[w];
    partials[blockIdx.x] = t;
  }
}

// ---------------------------------------------------------------------------
// Finish: deterministic reduction of block partials -> loss scalar
// ---------------------------------------------------------------------------
__global__ __launch_bounds__(256) void vq_finish(const float* __restrict__ partials,
                                                 int nparts,
                                                 float* __restrict__ loss_out) {
  __shared__ float s[256];
  float t = 0.f;
  for (int i = threadIdx.x; i < nparts; i += 256) t += partials[i];
  s[threadIdx.x] = t;
  __syncthreads();
  for (int stride = 128; stride > 0; stride >>= 1) {
    if (threadIdx.x < stride) s[threadIdx.x] += s[threadIdx.x + stride];
    __syncthreads();
  }
  if (threadIdx.x == 0)
    loss_out[0] = 1.25f * s[0] / (float)((long)N_ROWS * EMB_DIM);
}

// ---------------------------------------------------------------------------
extern "C" void kernel_launch(void* const* d_in, const int* in_sizes, int n_in,
                              void* d_out, int out_size, void* d_ws, size_t ws_size,
                              hipStream_t stream) {
  (void)in_sizes; (void)n_in; (void)out_size; (void)ws_size;
  const float* x   = (const float*)d_in[0];  // [64,4096,64] fp32
  const float* emb = (const float*)d_in[1];  // [512,64] fp32

  char* ws = (char*)d_ws;
  __bf16* ehi   = (__bf16*)(ws);             // 64 KB
  __bf16* elo   = (__bf16*)(ws + 65536);     // 64 KB
  float*  enorm = (float*)(ws + 131072);     // 2 KB
  float*  parts = (float*)(ws + 133120);     // 8 KB (2048 blocks)

  float* out_idx  = (float*)d_out;                     // N indices (as float)
  float* out_q    = out_idx + N_ROWS;                  // N*64 quantized
  float* out_loss = out_q + (size_t)N_ROWS * EMB_DIM;  // 1 scalar

  vq_prep<<<1, 512, 0, stream>>>(emb, ehi, elo, enorm);
  vq_main<<<N_ROWS / 128, 256, 0, stream>>>(x, emb, ehi, elo, enorm,
                                            out_idx, out_q, parts);
  vq_finish<<<1, 256, 0, stream>>>(parts, N_ROWS / 128, out_loss);
}
